// Seq2SeqAttention_10359461118587
// MI455X (gfx1250) — matrix operand order repaired and checked by reading, not compiled, not run on hardware
//
#include <hip/hip_runtime.h>
#include <hip/hip_bf16.h>
#include <cstdint>
#include <cstddef>

// ---------------------------------------------------------------------------
// Seq2Seq attention (encoder BiGRU + gate + attention decoder + NLL loss)
// CDNA5 / gfx1250: all large GEMMs run on v_wmma_f32_16x16x32_f16 (wave32).
// Each wave computes a 64x16 strip (4 M-tiles) so one B fragment is reused 4x,
// quartering weight-side L2 traffic in the dominant vocab-projection GEMM.
// ---------------------------------------------------------------------------

typedef _Float16 f16;
typedef __attribute__((ext_vector_type(16))) _Float16 v16h;
typedef __attribute__((ext_vector_type(8)))  float    v8f;

#define S_LEN 200
#define B_SZ  64
#define E_DIM 256
#define HIDD  512
#define HH_D  256
#define T_LEN 25
#define VOC   32000

__device__ __forceinline__ float sigmoidf_(float x) { return 1.0f / (1.0f + __expf(-x)); }

// A fragment: 16x32 f16 (M x K). ISA 7.12.2: lane = m%16 (both halves cover all M);
// VGPR v<4 holds k = kb + 2v + 8*half (+1); v>=4 holds k = kb + 16 + 2(v-4) + 8*half (+1).
__device__ __forceinline__ v16h load_a_frag(const f16* A, int lda, int row0, int kb, int lane) {
  int m = lane & 15, h = lane >> 4;
  const f16* row = A + (size_t)(row0 + m) * lda;
  v16h a;
#pragma unroll
  for (int v = 0; v < 8; ++v) {
    int k0 = kb + ((v < 4) ? (2 * v + 8 * h) : (16 + 2 * (v - 4) + 8 * h));
    a[2 * v]     = row[k0];
    a[2 * v + 1] = row[k0 + 1];
  }
  return a;
}

// B fragment: 32x16 f16 (K x N) sourced from row-major W[N x ldw] (B[k][n] = W[n][k]).
// Same k order as A: VGPR v<4 holds k = kb + 2v + 8*half (+1); v>=4 holds k = kb + 16 + 2(v-4) + 8*half (+1).
__device__ __forceinline__ v16h load_b_frag(const f16* W, int ldw, int wofs, int col0, int kb, int lane) {
  int n = lane & 15, h = lane >> 4;
  const f16* row = W + (size_t)(col0 + n) * ldw + wofs;
  v16h b;
#pragma unroll
  for (int v = 0; v < 8; ++v) {
    int k0 = kb + ((v < 4) ? (2 * v + 8 * h) : (16 + 2 * (v - 4) + 8 * h));
    b[2 * v]     = row[k0];
    b[2 * v + 1] = row[k0 + 1];
  }
  return b;
}

// ---------------------------------------------------------------------------
// Generic strip WMMA GEMM: C[M,N] = A[M,K](f16) @ W[N,K]^T(f16) (+bias+bias2+addC)
// One wave owns a 64x16 strip: 4 M-tiles reuse each B fragment. M % 64 == 0.
// ---------------------------------------------------------------------------
__global__ void wmma_gemm_kernel(const f16* __restrict__ A, int lda,
                                 const f16* __restrict__ W, int ldw, int wofs,
                                 const float* __restrict__ bias,
                                 const float* __restrict__ bias2,
                                 const float* __restrict__ addC,
                                 float* __restrict__ C, f16* __restrict__ Ch,
                                 int M, int N, int K) {
  int lane   = threadIdx.x & 31;
  int wave   = (blockIdx.x * blockDim.x + threadIdx.x) >> 5;
  int nwaves = (gridDim.x * blockDim.x) >> 5;
  int tilesN = N >> 4;
  int nstrip = (M >> 6) * tilesN;
  for (int t = wave; t < nstrip; t += nwaves) {
    int sm = t / tilesN, tn = t % tilesN;
    int m0 = sm * 64;
    v8f acc0 = {}, acc1 = {}, acc2 = {}, acc3 = {};
    for (int kb = 0; kb < K; kb += 32) {
      v16h b  = load_b_frag(W, ldw, wofs, tn * 16, kb, lane);
      v16h a0 = load_a_frag(A, lda, m0 +  0, kb, lane);
      v16h a1 = load_a_frag(A, lda, m0 + 16, kb, lane);
      v16h a2 = load_a_frag(A, lda, m0 + 32, kb, lane);
      v16h a3 = load_a_frag(A, lda, m0 + 48, kb, lane);
      acc0 = __builtin_amdgcn_wmma_f32_16x16x32_f16(false, a0, false, b, (short)0, acc0, false, false);
      acc1 = __builtin_amdgcn_wmma_f32_16x16x32_f16(false, a1, false, b, (short)0, acc1, false, false);
      acc2 = __builtin_amdgcn_wmma_f32_16x16x32_f16(false, a2, false, b, (short)0, acc2, false, false);
      acc3 = __builtin_amdgcn_wmma_f32_16x16x32_f16(false, a3, false, b, (short)0, acc3, false, false);
    }
    int n = tn * 16 + (lane & 15), h = lane >> 4;
#pragma unroll
    for (int q = 0; q < 4; ++q) {
      v8f acc = (q == 0) ? acc0 : (q == 1) ? acc1 : (q == 2) ? acc2 : acc3;
#pragma unroll
      for (int v = 0; v < 8; ++v) {
        int m = m0 + q * 16 + v + 8 * h;
        float val = acc[v];
        if (bias)  val += bias[n];
        if (bias2) val += bias2[n];
        if (addC)  val += addC[(size_t)m * N + n];
        if (C)  C[(size_t)m * N + n] = val;
        if (Ch) Ch[(size_t)m * N + n] = (f16)val;
      }
    }
  }
}

// Gate GEMM: g = sigmoid(states_h @ W1^T + term2[m%64]); states *= g  (N=K=512)
__global__ void gate_gemm_kernel(const f16* __restrict__ A, const f16* __restrict__ W1h,
                                 const float* __restrict__ term2, float* __restrict__ states) {
  int lane   = threadIdx.x & 31;
  int wave   = (blockIdx.x * blockDim.x + threadIdx.x) >> 5;
  int nwaves = (gridDim.x * blockDim.x) >> 5;
  const int tilesN = HIDD / 16;                    // 32
  const int nstrip = (S_LEN * B_SZ / 64) * tilesN; // 200*32
  for (int t = wave; t < nstrip; t += nwaves) {
    int sm = t / tilesN, tn = t % tilesN;
    int m0 = sm * 64;
    v8f acc0 = {}, acc1 = {}, acc2 = {}, acc3 = {};
    for (int kb = 0; kb < HIDD; kb += 32) {
      v16h b  = load_b_frag(W1h, HIDD, 0, tn * 16, kb, lane);
      v16h a0 = load_a_frag(A, HIDD, m0 +  0, kb, lane);
      v16h a1 = load_a_frag(A, HIDD, m0 + 16, kb, lane);
      v16h a2 = load_a_frag(A, HIDD, m0 + 32, kb, lane);
      v16h a3 = load_a_frag(A, HIDD, m0 + 48, kb, lane);
      acc0 = __builtin_amdgcn_wmma_f32_16x16x32_f16(false, a0, false, b, (short)0, acc0, false, false);
      acc1 = __builtin_amdgcn_wmma_f32_16x16x32_f16(false, a1, false, b, (short)0, acc1, false, false);
      acc2 = __builtin_amdgcn_wmma_f32_16x16x32_f16(false, a2, false, b, (short)0, acc2, false, false);
      acc3 = __builtin_amdgcn_wmma_f32_16x16x32_f16(false, a3, false, b, (short)0, acc3, false, false);
    }
    int n = tn * 16 + (lane & 15), h = lane >> 4;
#pragma unroll
    for (int q = 0; q < 4; ++q) {
      v8f acc = (q == 0) ? acc0 : (q == 1) ? acc1 : (q == 2) ? acc2 : acc3;
#pragma unroll
      for (int v = 0; v < 8; ++v) {
        int m = m0 + q * 16 + v + 8 * h;
        float g = sigmoidf_(acc[v] + term2[(m & 63) * HIDD + n]);
        states[(size_t)m * HIDD + n] *= g;
      }
    }
  }
}

// ---------------------------------------------------------------------------
// Encoder recurrence: one block per direction, 32 waves. h kept in LDS as f16
// for WMMA A-fragments; gh = h @ w_hh^T each step via WMMA. Each wave owns a
// 64x16 strip (full M) so the w_hh B fragment is loaded once per strip.
// ---------------------------------------------------------------------------
__global__ __launch_bounds__(1024) void encoder_rnn_kernel(
    const f16* __restrict__ gif, const f16* __restrict__ gib,
    const f16* __restrict__ Whhf, const f16* __restrict__ Whhb,
    const float* __restrict__ bhhf, const float* __restrict__ bhhb,
    float* __restrict__ hbuf, float* __restrict__ ghbuf,
    float* __restrict__ states, f16* __restrict__ states_h) {
  __shared__ f16 h_h[B_SZ * HH_D];  // 32 KB
  int dir = blockIdx.x;
  const f16* gi   = dir ? gib  : gif;
  const f16* Whh  = dir ? Whhb : Whhf;
  const float* bh = dir ? bhhb : bhhf;
  float* hb = hbuf  + (size_t)dir * B_SZ * HH_D;
  float* gh = ghbuf + (size_t)dir * B_SZ * 3 * HH_D;
  int tid = threadIdx.x, lane = tid & 31, wv = tid >> 5;

  for (int i = tid; i < B_SZ * HH_D; i += blockDim.x) { hb[i] = 0.f; h_h[i] = (f16)0.f; }
  __syncthreads();

  for (int step = 0; step < S_LEN; ++step) {
    int ti = dir ? (S_LEN - 1 - step) : step;  // time index of x processed this step
    // gh = h @ Whh^T : M=64, N=768, K=256 -> 48 strips of 64x16
    for (int tn = wv; tn < 48; tn += 32) {
      v8f acc0 = {}, acc1 = {}, acc2 = {}, acc3 = {};
      for (int kb = 0; kb < HH_D; kb += 32) {
        v16h b  = load_b_frag(Whh, HH_D, 0, tn * 16, kb, lane);
        v16h a0 = load_a_frag(h_h, HH_D,  0, kb, lane);
        v16h a1 = load_a_frag(h_h, HH_D, 16, kb, lane);
        v16h a2 = load_a_frag(h_h, HH_D, 32, kb, lane);
        v16h a3 = load_a_frag(h_h, HH_D, 48, kb, lane);
        acc0 = __builtin_amdgcn_wmma_f32_16x16x32_f16(false, a0, false, b, (short)0, acc0, false, false);
        acc1 = __builtin_amdgcn_wmma_f32_16x16x32_f16(false, a1, false, b, (short)0, acc1, false, false);
        acc2 = __builtin_amdgcn_wmma_f32_16x16x32_f16(false, a2, false, b, (short)0, acc2, false, false);
        acc3 = __builtin_amdgcn_wmma_f32_16x16x32_f16(false, a3, false, b, (short)0, acc3, false, false);
      }
      int n = tn * 16 + (lane & 15), h2 = lane >> 4;
#pragma unroll
      for (int q = 0; q < 4; ++q) {
        v8f acc = (q == 0) ? acc0 : (q == 1) ? acc1 : (q == 2) ? acc2 : acc3;
#pragma unroll
        for (int v = 0; v < 8; ++v) gh[(q * 16 + v + 8 * h2) * (3 * HH_D) + n] = acc[v];
      }
    }
    __threadfence_block();
    __syncthreads();
    // elementwise gates
    for (int i = tid; i < B_SZ * HH_D; i += blockDim.x) {
      int b = i >> 8, j = i & 255;
      size_t gbase = ((size_t)ti * B_SZ + b) * (3 * HH_D);
      float ir  = (float)gi[gbase + j];
      float iz  = (float)gi[gbase + HH_D + j];
      float in_ = (float)gi[gbase + 2 * HH_D + j];
      float hr = gh[b * 3 * HH_D + j]            + bh[j];
      float hz = gh[b * 3 * HH_D + HH_D + j]     + bh[HH_D + j];
      float hn = gh[b * 3 * HH_D + 2 * HH_D + j] + bh[2 * HH_D + j];
      float r = sigmoidf_(ir + hr), z = sigmoidf_(iz + hz);
      float nn = tanhf(in_ + r * hn);
      float hnew = (1.f - z) * nn + z * hb[i];
      hb[i] = hnew;
      h_h[i] = (f16)hnew;
      size_t so = ((size_t)ti * B_SZ + b) * HIDD + dir * HH_D + j;
      states[so]   = hnew;
      states_h[so] = (f16)hnew;
    }
    __threadfence_block();
    __syncthreads();
  }
  // final h stays in hbuf; flat layout [2][64][256] == enc_hidden.reshape(B,HID) flat.
}

// ---------------------------------------------------------------------------
// Small helper kernels
// ---------------------------------------------------------------------------
__global__ void f32_to_f16_kernel(const float* __restrict__ src, f16* __restrict__ dst, int n) {
  int i = blockIdx.x * blockDim.x + threadIdx.x, st = gridDim.x * blockDim.x;
  for (; i < n; i += st) dst[i] = (f16)src[i];
}

__global__ void embed_src_kernel(const int* __restrict__ src, const float* __restrict__ emb,
                                 f16* __restrict__ xh) {
  int i = blockIdx.x * blockDim.x + threadIdx.x, st = gridDim.x * blockDim.x;
  const int n = S_LEN * B_SZ * E_DIM;  // x[t,b] = emb[src_flat[t*B+b]] (faithful .view)
  for (; i < n; i += st) {
    int row = i >> 8, col = i & 255;
    xh[i] = (f16)emb[(size_t)src[row] * E_DIM + col];
  }
}

__global__ void embed_trg_kernel(const int* __restrict__ trg, const float* __restrict__ emb,
                                 f16* __restrict__ eh) {
  int i = blockIdx.x * blockDim.x + threadIdx.x, st = gridDim.x * blockDim.x;
  const int n = (T_LEN - 1) * B_SZ * E_DIM;  // row m = t*64+b -> token trg[b*T + t]
  for (; i < n; i += st) {
    int row = i >> 8, col = i & 255;
    int t = row >> 6, b = row & 63;
    eh[i] = (f16)emb[(size_t)trg[b * T_LEN + t] * E_DIM + col];
  }
}

__global__ void attn1_kernel(const float* __restrict__ states, const float* __restrict__ Wa1,
                             const float* __restrict__ ba1, float* __restrict__ attn1) {
  __shared__ float red[128];
  int m = blockIdx.x, tid = threadIdx.x;
  float s = 0.f;
  for (int j = tid; j < HIDD; j += 128) s += states[(size_t)m * HIDD + j] * Wa1[j];
  red[tid] = s; __syncthreads();
  for (int off = 64; off; off >>= 1) { if (tid < off) red[tid] += red[tid + off]; __syncthreads(); }
  if (!tid) attn1[m] = red[0] + ba1[0];
}

__global__ void dec_a2_kernel(const float* __restrict__ h, const float* __restrict__ Wa2,
                              const float* __restrict__ ba2, float* __restrict__ a2) {
  __shared__ float red[256];
  int b = blockIdx.x, tid = threadIdx.x;
  float s = 0.f;
  for (int j = tid; j < HIDD; j += 256) s += h[b * HIDD + j] * Wa2[j];
  red[tid] = s; __syncthreads();
  for (int off = 128; off; off >>= 1) { if (tid < off) red[tid] += red[tid + off]; __syncthreads(); }
  if (!tid) a2[b] = red[0] + ba2[0];
}

// Faithful .view(batch,-1): softmax over flat chunks of 200 of the s-major array.
__global__ void dec_softmax_kernel(const float* __restrict__ attn1, const float* __restrict__ a2,
                                   float* __restrict__ aw) {
  __shared__ float red[256];
  __shared__ float vals[S_LEN];
  int g = blockIdx.x, tid = threadIdx.x;
  int base = g * S_LEN;
  float mx = -1e30f;
  for (int i = tid; i < S_LEN; i += 256) {
    float v = attn1[base + i] + a2[(base + i) & 63];
    vals[i] = v; mx = fmaxf(mx, v);
  }
  red[tid] = mx; __syncthreads();
  for (int off = 128; off; off >>= 1) { if (tid < off) red[tid] = fmaxf(red[tid], red[tid + off]); __syncthreads(); }
  mx = red[0]; __syncthreads();
  float sm = 0.f;
  for (int i = tid; i < S_LEN; i += 256) { float e = __expf(vals[i] - mx); vals[i] = e; sm += e; }
  red[tid] = sm; __syncthreads();
  for (int off = 128; off; off >>= 1) { if (tid < off) red[tid] += red[tid + off]; __syncthreads(); }
  float inv = 1.0f / red[0];
  for (int i = tid; i < S_LEN; i += 256) aw[base + i] = vals[i] * inv;
}

__global__ void dec_context_kernel(const float* __restrict__ aw, const float* __restrict__ states,
                                   float* __restrict__ c, f16* __restrict__ ch) {
  int b = blockIdx.x, tid = threadIdx.x;
  for (int j = tid; j < HIDD; j += 256) {
    float acc = 0.f;
    for (int s = 0; s < S_LEN; ++s)
      acc += aw[s * B_SZ + b] * states[((size_t)s * B_SZ + b) * HIDD + j];
    c[b * HIDD + j]  = acc;
    ch[b * HIDD + j] = (f16)acc;
  }
}

__global__ void dec_gates_kernel(const float* __restrict__ gi, const float* __restrict__ gh,
                                 float* __restrict__ h, f16* __restrict__ hh) {
  int idx = blockIdx.x * blockDim.x + threadIdx.x;
  if (idx >= B_SZ * HIDD) return;
  int b = idx >> 9, j = idx & 511;
  const float* gib = gi + b * 3 * HIDD;
  const float* ghb = gh + b * 3 * HIDD;
  float r  = sigmoidf_(gib[j] + ghb[j]);
  float z  = sigmoidf_(gib[HIDD + j] + ghb[HIDD + j]);
  float nn = tanhf(gib[2 * HIDD + j] + r * ghb[2 * HIDD + j]);
  float hnew = (1.f - z) * nn + z * h[idx];
  h[idx]  = hnew;
  hh[idx] = (f16)hnew;
}

__global__ void loss_rows_kernel(const float* __restrict__ logits, const int* __restrict__ trg,
                                 int tstep, float* __restrict__ loss) {
  __shared__ float red[256];
  int b = blockIdx.x, tid = threadIdx.x;
  const float* row = logits + (size_t)b * VOC;
  float mx = -1e30f;
  for (int i = tid; i < VOC; i += 256) mx = fmaxf(mx, row[i]);
  red[tid] = mx; __syncthreads();
  for (int off = 128; off; off >>= 1) { if (tid < off) red[tid] = fmaxf(red[tid], red[tid + off]); __syncthreads(); }
  mx = red[0]; __syncthreads();
  float sm = 0.f;
  for (int i = tid; i < VOC; i += 256) sm += __expf(row[i] - mx);
  red[tid] = sm; __syncthreads();
  for (int off = 128; off; off >>= 1) { if (tid < off) red[tid] += red[tid + off]; __syncthreads(); }
  if (!tid) {
    int tgt = trg[b * T_LEN + tstep + 1];
    atomicAdd(loss, (mx + __logf(red[0])) - row[tgt]);
  }
}

__global__ void init_loss_kernel(float* loss) { if (!threadIdx.x && !blockIdx.x) *loss = 0.f; }
__global__ void finalize_kernel(const float* loss, float* out) {
  if (!threadIdx.x && !blockIdx.x) out[0] = *loss / (float)(B_SZ * T_LEN);
}

// ---------------------------------------------------------------------------
extern "C" void kernel_launch(void* const* d_in, const int* in_sizes, int n_in,
                              void* d_out, int out_size, void* d_ws, size_t ws_size,
                              hipStream_t stream) {
  const int*   src    = (const int*)d_in[0];
  const int*   trg    = (const int*)d_in[1];
  const float* emb    = (const float*)d_in[2];
  const float* w_ih_f = (const float*)d_in[3];
  const float* w_hh_f = (const float*)d_in[4];
  const float* b_ih_f = (const float*)d_in[5];
  const float* b_hh_f = (const float*)d_in[6];
  const float* w_ih_b = (const float*)d_in[7];
  const float* w_hh_b = (const float*)d_in[8];
  const float* b_ih_b = (const float*)d_in[9];
  const float* b_hh_b = (const float*)d_in[10];
  const float* W1     = (const float*)d_in[11];
  const float* b1     = (const float*)d_in[12];
  const float* W2     = (const float*)d_in[13];
  const float* b2     = (const float*)d_in[14];
  const float* w_ih_d = (const float*)d_in[15];
  const float* w_hh_d = (const float*)d_in[16];
  const float* b_ih_d = (const float*)d_in[17];
  const float* b_hh_d = (const float*)d_in[18];
  const float* Wv     = (const float*)d_in[19];
  const float* bv     = (const float*)d_in[20];
  const float* Wa1    = (const float*)d_in[21];
  const float* ba1    = (const float*)d_in[22];
  const float* Wa2    = (const float*)d_in[23];
  const float* ba2    = (const float*)d_in[24];

  char* ws = (char*)d_ws;
  size_t off = 0;
  auto carve = [&](size_t bytes) -> char* {
    char* p = ws + off;
    off = (off + bytes + 255) & ~(size_t)255;
    return p;
  };

  // f16 scratch
  f16* Wihf_h = (f16*)carve((size_t)768 * 256 * 2);
  f16* Whhf_h = (f16*)carve((size_t)768 * 256 * 2);
  f16* Wihb_h = (f16*)carve((size_t)768 * 256 * 2);
  f16* Whhb_h = (f16*)carve((size_t)768 * 256 * 2);
  f16* W1_h   = (f16*)carve((size_t)512 * 512 * 2);
  f16* W2_h   = (f16*)carve((size_t)512 * 512 * 2);
  f16* Wihd_h = (f16*)carve((size_t)1536 * 768 * 2);
  f16* Whhd_h = (f16*)carve((size_t)1536 * 512 * 2);
  f16* Wv_h   = (f16*)carve((size_t)VOC * 512 * 2);
  f16* x_h    = (f16*)carve((size_t)12800 * 256 * 2);
  f16* gif_h  = (f16*)carve((size_t)12800 * 768 * 2);
  f16* gib_h  = (f16*)carve((size_t)12800 * 768 * 2);
  f16* st_h   = (f16*)carve((size_t)12800 * 512 * 2);
  f16* e_h    = (f16*)carve((size_t)1536 * 256 * 2);
  f16* hdec_h = (f16*)carve((size_t)64 * 512 * 2);
  f16* c_h    = (f16*)carve((size_t)64 * 512 * 2);
  // f32 scratch
  float* states  = (float*)carve((size_t)12800 * 512 * 4);
  float* gi_e    = (float*)carve((size_t)1536 * 1536 * 4);
  float* term2   = (float*)carve((size_t)64 * 512 * 4);
  float* attn1   = (float*)carve((size_t)12800 * 4);
  float* a2      = (float*)carve((size_t)64 * 4);
  float* aw      = (float*)carve((size_t)12800 * 4);
  float* cbuf    = (float*)carve((size_t)64 * 512 * 4);
  float* gi_full = (float*)carve((size_t)64 * 1536 * 4);
  float* gh_full = (float*)carve((size_t)64 * 1536 * 4);
  float* logits  = (float*)carve((size_t)64 * VOC * 4);
  float* hbuf    = (float*)carve((size_t)2 * 64 * 256 * 4);  // also decoder h (flat view match)
  float* ghbuf   = (float*)carve((size_t)2 * 64 * 768 * 4);
  float* loss    = (float*)carve(4);

  if (off > ws_size) {  // workspace too small: emit defined output, do nothing else
    init_loss_kernel<<<1, 1, 0, stream>>>((float*)d_out);
    return;
  }

  auto cvt = [&](const float* s, f16* d, size_t n) {
    int blocks = (int)((n + 255) / 256); if (blocks > 2048) blocks = 2048;
    f32_to_f16_kernel<<<blocks, 256, 0, stream>>>(s, d, (int)n);
  };
  auto gemm = [&](const f16* A, int lda, const f16* W, int ldw, int wofs,
                  const float* bp, const float* bp2, const float* addC,
                  float* C, f16* Ch, int M, int N, int K) {
    int strips = (M / 64) * (N / 16);
    int blocks = (strips + 7) / 8; if (blocks > 4096) blocks = 4096; if (blocks < 1) blocks = 1;
    wmma_gemm_kernel<<<blocks, 256, 0, stream>>>(A, lda, W, ldw, wofs, bp, bp2, addC, C, Ch, M, N, K);
  };

  init_loss_kernel<<<1, 1, 0, stream>>>(loss);

  // weights -> f16 (L2-resident working set)
  cvt(w_ih_f, Wihf_h, (size_t)768 * 256);
  cvt(w_hh_f, Whhf_h, (size_t)768 * 256);
  cvt(w_ih_b, Wihb_h, (size_t)768 * 256);
  cvt(w_hh_b, Whhb_h, (size_t)768 * 256);
  cvt(W1, W1_h, (size_t)512 * 512);
  cvt(W2, W2_h, (size_t)512 * 512);
  cvt(w_ih_d, Wihd_h, (size_t)1536 * 768);
  cvt(w_hh_d, Whhd_h, (size_t)1536 * 512);
  cvt(Wv, Wv_h, (size_t)VOC * 512);

  // encoder: hoisted input projections, then persistent recurrence
  embed_src_kernel<<<2048, 256, 0, stream>>>(src, emb, x_h);
  gemm(x_h, 256, Wihf_h, 256, 0, b_ih_f, nullptr, nullptr, nullptr, gif_h, 12800, 768, 256);
  gemm(x_h, 256, Wihb_h, 256, 0, b_ih_b, nullptr, nullptr, nullptr, gib_h, 12800, 768, 256);
  encoder_rnn_kernel<<<2, 1024, 0, stream>>>(gif_h, gib_h, Whhf_h, Whhb_h, b_hh_f, b_hh_b,
                                             hbuf, ghbuf, states, st_h);

  // gate: term2 = enc_h_flat @ W2^T + b2 + b1 (flat hbuf == reshape(B,HID)), then big gated GEMM
  f32_to_f16_kernel<<<128, 256, 0, stream>>>(hbuf, hdec_h, 64 * 512);
  gemm(hdec_h, 512, W2_h, 512, 0, b1, b2, nullptr, term2, nullptr, 64, 512, 512);
  {
    int strips = 200 * 32;
    int blocks = (strips + 7) / 8; if (blocks > 4096) blocks = 4096;
    gate_gemm_kernel<<<blocks, 256, 0, stream>>>(st_h, W1_h, term2, states);
  }
  attn1_kernel<<<12800, 128, 0, stream>>>(states, Wa1, ba1, attn1);

  // decoder: hoisted embedding projection, then 24 sequential steps
  embed_trg_kernel<<<1536, 256, 0, stream>>>(trg, emb, e_h);
  gemm(e_h, 256, Wihd_h, 768, 0, b_ih_d, nullptr, nullptr, gi_e, nullptr, 1536, 1536, 256);

  for (int t = 0; t < T_LEN - 1; ++t) {
    dec_a2_kernel<<<64, 256, 0, stream>>>(hbuf, Wa2, ba2, a2);
    dec_softmax_kernel<<<64, 256, 0, stream>>>(attn1, a2, aw);
    dec_context_kernel<<<64, 256, 0, stream>>>(aw, states, cbuf, c_h);
    // gi = gi_e[t] + c @ w_ih_d[:,E:]^T ; gh = h @ w_hh_d^T + b_hh_d
    gemm(c_h, 512, Wihd_h, 768, 256, nullptr, nullptr, gi_e + (size_t)t * 64 * 1536,
         gi_full, nullptr, 64, 1536, 512);
    gemm(hdec_h, 512, Whhd_h, 512, 0, b_hh_d, nullptr, nullptr, gh_full, nullptr, 64, 1536, 512);
    dec_gates_kernel<<<128, 256, 0, stream>>>(gi_full, gh_full, hbuf, hdec_h);
    // vocab projection: M=64, N=32000, K=512 -> 2000 strips across the device
    gemm(hdec_h, 512, Wv_h, 512, 0, bv, nullptr, nullptr, logits, nullptr, 64, VOC, 512);
    loss_rows_kernel<<<64, 256, 0, stream>>>(logits, trg, t, loss);
  }

  finalize_kernel<<<1, 1, 0, stream>>>(loss, (float*)d_out);
}
